// AttentionDownscaler_72181220376700
// MI455X (gfx1250) — matrix operand order repaired and checked by reading, not compiled, not run on hardware
//
#include <hip/hip_runtime.h>
#include <hip/hip_fp16.h>

typedef __attribute__((ext_vector_type(16))) _Float16 v16h;
typedef __attribute__((ext_vector_type(8)))  _Float16 v8h;
typedef __attribute__((ext_vector_type(8)))  float    v8f;

#define DM      14
#define NHEAD   2
#define DH      7
#define NLAYER  3
#define FFD     64
#define LCH     7500
#define TPAD    7552           // LCH padded to multiple of 64
#define STRIDE1 5500
#define KW      500
#define SEQ1    30000
#define BB      2
#define NCHUNK1 5
#define NCOL    37500          // NCHUNK1 * LCH
#define MROWS   75000          // BB * NCOL
#define MPAD    75008          // MROWS padded to multiple of 16
#define L2LEN   15000
#define EPSF    1e-5f
#define INVSQRT7 0.3779644730092272f

// ---------------- helpers ----------------

// Fast reciprocal: single v_rcp_f32 (~1 ulp) instead of the IEEE
// div_scale/rcp/fma Newton-Raphson chain.
__device__ __forceinline__ float frcp(float x){
    return __builtin_amdgcn_rcpf(x);
}

// Branch-free erf (Abramowitz-Stegun 7.1.26, |err| < 1.5e-7) on the fast
// v_exp_f32 / v_rcp_f32 paths -- no divergent saveexec branches.
__device__ __forceinline__ float erf_fast(float x){
    float ax = fabsf(x);
    float t  = frcp(fmaf(0.3275911f, ax, 1.0f));
    float p  = t * fmaf(t, fmaf(t, fmaf(t, fmaf(t, 1.061405429f, -1.453152027f),
                                        1.421413741f), -0.284496736f), 0.254829592f);
    float r  = 1.0f - p * __expf(-ax * ax);
    return copysignf(r, x);
}
__device__ __forceinline__ float gelu_exact(float x){
    return 0.5f * x * (1.0f + erf_fast(x * 0.7071067811865475f));
}

// reductions across the 16 lanes of each half-wave (wave32); xor masks < 16
// keep halves independent, matching the 16x16 f32 C layout
// (row m = r + 8*(lane>=16), col n = lane&15).
__device__ __forceinline__ float hred_max(float v){
    v = fmaxf(v, __shfl_xor(v, 1, 32));
    v = fmaxf(v, __shfl_xor(v, 2, 32));
    v = fmaxf(v, __shfl_xor(v, 4, 32));
    v = fmaxf(v, __shfl_xor(v, 8, 32));
    return v;
}
__device__ __forceinline__ float hred_sum(float v){
    v += __shfl_xor(v, 1, 32);
    v += __shfl_xor(v, 2, 32);
    v += __shfl_xor(v, 4, 32);
    v += __shfl_xor(v, 8, 32);
    return v;
}

// A-fragment (16x32 f16, row = lane&15): halves 0-7  = K[hi*8 .. hi*8+7],
//                                        halves 8-15 = K[16+hi*8 .. 16+hi*8+7]
__device__ __forceinline__ v16h load_a_frag(const _Float16* row32, int hi){
    v8h lo = *(const v8h*)(row32 + hi * 8);
    v8h hv = *(const v8h*)(row32 + 16 + hi * 8);
    return __builtin_shufflevector(lo, hv, 0,1,2,3,4,5,6,7,8,9,10,11,12,13,14,15);
}
// B-fragment (32x16 f16, col = lane&15): halves 0-7  = K[hi*8 .. hi*8+7],
//                                        halves 8-15 = K[16+hi*8 .. 16+hi*8+7]
__device__ __forceinline__ v16h load_b_frag(const _Float16* col32, int hi){
    v8h lo = *(const v8h*)(col32 + hi * 8);
    v8h hv = *(const v8h*)(col32 + 16 + hi * 8);
    return __builtin_shufflevector(lo, hv, 0,1,2,3,4,5,6,7,8,9,10,11,12,13,14,15);
}

// ---------------- positional encoding ----------------

__global__ void pe_kernel(float* __restrict__ pe){
    int i = blockIdx.x * blockDim.x + threadIdx.x;
    if (i >= LCH * DM) return;
    int l = i / DM, d = i % DM;
    int pair = d >> 1;
    float div = __expf((float)(2 * pair) * (-__logf(10000.0f) / (float)DM));
    float ang = (float)l * div;
    pe[i] = (d & 1) ? __cosf(ang) : __sinf(ang);
}

// ---------------- stage 1: fold chunks into columns ----------------
__global__ void fold_kernel(const float* __restrict__ g, const float* __restrict__ pe,
                            float* __restrict__ x1){
    int idx = blockIdx.x * blockDim.x + threadIdx.x;
    if (idx >= BB * NCHUNK1 * LCH * DM) return;
    int d = idx % DM;
    int r = idx / DM;
    int t = r % LCH; r /= LCH;
    int c = r % NCHUNK1;
    int b = r / NCHUNK1;
    x1[(((long)b * NCOL) + c * LCH + t) * DM + d] =
        g[(((long)b * SEQ1) + c * STRIDE1 + t) * DM + d] + pe[t * DM + d];
}

// ---------------- weight packing (once): w1h[3][64][32], w2h[3][16][64] -------
__global__ void packw_kernel(const float* __restrict__ w1, const float* __restrict__ w2,
                             _Float16* __restrict__ w1h, _Float16* __restrict__ w2h){
    int idx = blockIdx.x * blockDim.x + threadIdx.x;
    if (idx < NLAYER * FFD * 32){
        int k = idx % 32, f = (idx / 32) % FFD, l = idx / (32 * FFD);
        w1h[idx] = (k < DM) ? (_Float16)w1[((long)l * FFD + f) * DM + k] : (_Float16)0.f;
    } else if (idx < NLAYER * FFD * 32 + NLAYER * 16 * FFD){
        int i2 = idx - NLAYER * FFD * 32;
        int k = i2 % FFD, nn = (i2 / FFD) % 16, l = i2 / (FFD * 16);
        w2h[i2] = (nn < DM) ? (_Float16)w2[((long)l * DM + nn) * FFD + k] : (_Float16)0.f;
    }
}

// ---------------- activation packing per layer: xh[75008][32] f16 -------------
__global__ void packx_kernel(const float* __restrict__ x, _Float16* __restrict__ xh){
    int idx = blockIdx.x * blockDim.x + threadIdx.x;
    if (idx >= MPAD * 32) return;
    int k = idx % 32;
    int row = idx / 32;
    xh[idx] = (row < MROWS && k < DM) ? (_Float16)x[(long)row * DM + k] : (_Float16)0.f;
}

// ---------------- stage 1: attention over S=2 + out-proj + LN1 ----------------
__global__ void enc_attn_kernel(float* __restrict__ x,
    const float* __restrict__ win, const float* __restrict__ bin,
    const float* __restrict__ wout, const float* __restrict__ bout,
    const float* __restrict__ lnw, const float* __restrict__ lnb)
{
    int nidx = blockIdx.x * blockDim.x + threadIdx.x;
    if (nidx >= NCOL) return;
    float x0[DM], x1r[DM];
    #pragma unroll
    for (int d = 0; d < DM; ++d){
        x0[d]  = x[((long)nidx) * DM + d];
        x1r[d] = x[((long)NCOL + nidx) * DM + d];
    }
    float y0[DM], y1[DM];
    #pragma unroll
    for (int e = 0; e < DM; ++e){ y0[e] = bout[e]; y1[e] = bout[e]; }

    for (int h = 0; h < NHEAD; ++h){
        float q0[DH], k0[DH], v0[DH], q1[DH], k1[DH], v1[DH];
        #pragma unroll
        for (int d = 0; d < DH; ++d){
            int oq = h * DH + d, ok = DM + h * DH + d, ov = 2 * DM + h * DH + d;
            float aq0 = bin[oq], ak0 = bin[ok], av0 = bin[ov];
            float aq1 = aq0, ak1 = ak0, av1 = av0;
            #pragma unroll
            for (int e = 0; e < DM; ++e){
                float wq = win[oq * DM + e], wk = win[ok * DM + e], wv = win[ov * DM + e];
                aq0 = fmaf(wq, x0[e], aq0);  aq1 = fmaf(wq, x1r[e], aq1);
                ak0 = fmaf(wk, x0[e], ak0);  ak1 = fmaf(wk, x1r[e], ak1);
                av0 = fmaf(wv, x0[e], av0);  av1 = fmaf(wv, x1r[e], av1);
            }
            q0[d] = aq0; q1[d] = aq1; k0[d] = ak0; k1[d] = ak1; v0[d] = av0; v1[d] = av1;
        }
        float s00 = 0.f, s01 = 0.f, s10 = 0.f, s11 = 0.f;
        #pragma unroll
        for (int d = 0; d < DH; ++d){
            s00 = fmaf(q0[d], k0[d], s00);  s01 = fmaf(q0[d], k1[d], s01);
            s10 = fmaf(q1[d], k0[d], s10);  s11 = fmaf(q1[d], k1[d], s11);
        }
        s00 *= INVSQRT7; s01 *= INVSQRT7; s10 *= INVSQRT7; s11 *= INVSQRT7;
        float m0r = fmaxf(s00, s01), m1r = fmaxf(s10, s11);
        float e00 = __expf(s00 - m0r), e01 = __expf(s01 - m0r);
        float e10 = __expf(s10 - m1r), e11 = __expf(s11 - m1r);
        float r0i = frcp(e00 + e01), r1i = frcp(e10 + e11);
        float p00 = e00 * r0i, p01 = e01 * r0i;
        float p10 = e10 * r1i, p11 = e11 * r1i;
        #pragma unroll
        for (int d = 0; d < DH; ++d){
            float o0 = p00 * v0[d] + p01 * v1[d];
            float o1 = p10 * v0[d] + p11 * v1[d];
            #pragma unroll
            for (int e = 0; e < DM; ++e){
                float w = wout[e * DM + h * DH + d];
                y0[e] = fmaf(w, o0, y0[e]);
                y1[e] = fmaf(w, o1, y1[e]);
            }
        }
    }
    float r0[DM], r1[DM], m0 = 0.f, m1 = 0.f;
    #pragma unroll
    for (int d = 0; d < DM; ++d){
        r0[d] = x0[d] + y0[d]; r1[d] = x1r[d] + y1[d];
        m0 += r0[d]; m1 += r1[d];
    }
    m0 *= (1.0f / DM); m1 *= (1.0f / DM);
    float v0s = 0.f, v1s = 0.f;
    #pragma unroll
    for (int d = 0; d < DM; ++d){
        float a = r0[d] - m0, bq = r1[d] - m1;
        v0s = fmaf(a, a, v0s); v1s = fmaf(bq, bq, v1s);
    }
    float i0 = rsqrtf(v0s * (1.0f / DM) + EPSF);
    float i1 = rsqrtf(v1s * (1.0f / DM) + EPSF);
    #pragma unroll
    for (int d = 0; d < DM; ++d){
        x[((long)nidx) * DM + d]        = (r0[d] - m0) * i0 * lnw[d] + lnb[d];
        x[((long)NCOL + nidx) * DM + d] = (r1[d] - m1) * i1 * lnw[d] + lnb[d];
    }
}

// ---------------- stage 1: fused FF (WMMA) + residual + LN2 ----------------
__global__ void __launch_bounds__(32) enc_ff_kernel(
    const _Float16* __restrict__ xh,   // [MPAD][32] packed f16 (post-LN1)
    float* __restrict__ x,             // [MROWS][DM] f32 (in/out)
    const _Float16* __restrict__ w1h,  // [FFD][32]
    const float* __restrict__ b1,
    const _Float16* __restrict__ w2h,  // [16][64]
    const float* __restrict__ b2,
    const float* __restrict__ lnw, const float* __restrict__ lnb)
{
    __shared__ __align__(16) _Float16 hbuf[16][FFD];
    const int lane = threadIdx.x;
    const int n = lane & 15, hi = lane >> 4;
    const int m0 = blockIdx.x * 16;

    v16h a = load_a_frag(xh + (long)(m0 + (lane & 15)) * 32, hi);
    v8f c[4];
    #pragma unroll
    for (int nt = 0; nt < 4; ++nt){
        v16h bfr = load_b_frag(w1h + (long)(nt * 16 + n) * 32, hi);
        v8f z = {};
        c[nt] = __builtin_amdgcn_wmma_f32_16x16x32_f16(false, a, false, bfr,
                                                       (short)0, z, false, false);
    }
    #pragma unroll
    for (int nt = 0; nt < 4; ++nt){
        int f = nt * 16 + n;
        #pragma unroll
        for (int r = 0; r < 8; ++r){
            float h = gelu_exact(c[nt][r] + b1[f]);
            hbuf[r + 8 * hi][f] = (_Float16)h;
        }
    }
    __syncthreads();

    v8f c2 = {};
    #pragma unroll
    for (int ks = 0; ks < 2; ++ks){
        v16h a2  = load_a_frag(&hbuf[lane & 15][32 * ks], hi);
        v16h b2f = load_b_frag(w2h + (long)n * FFD + 32 * ks, hi);
        c2 = __builtin_amdgcn_wmma_f32_16x16x32_f16(false, a2, false, b2f,
                                                    (short)0, c2, false, false);
    }
    const int nc = (n < DM) ? n : (DM - 1);
    const float b2v = b2[nc], lwv = lnw[nc], lbv = lnb[nc];
    #pragma unroll
    for (int r = 0; r < 8; ++r){
        int m = m0 + r + 8 * hi;
        int mc = (m < MROWS) ? m : (MROWS - 1);
        float xres = x[(long)mc * DM + nc];                 // unconditional load
        float resid = (n < DM) ? (xres + c2[r] + b2v) : 0.f;
        float mean = hred_sum(resid) * (1.0f / DM);
        float dv = (n < DM) ? (resid - mean) : 0.f;
        float var = hred_sum(dv * dv) * (1.0f / DM);
        if (m < MROWS && n < DM)
            x[(long)m * DM + n] = dv * rsqrtf(var + EPSF) * lwv + lbv;
    }
}

// ---------------- stage 1: overlap-add back to sequence ----------------
__global__ void unfold_kernel(const float* __restrict__ x1, float* __restrict__ out1){
    int idx = blockIdx.x * blockDim.x + threadIdx.x;
    if (idx >= BB * SEQ1 * DM) return;
    int d = idx % DM;
    int r = idx / DM;
    int s = r % SEQ1;
    int b = r / SEQ1;
    float acc = 0.f; int cnt = 0;
    #pragma unroll
    for (int c = 0; c < NCHUNK1; ++c){
        int t = s - c * STRIDE1;
        if (t >= 0 && t < LCH){
            acc += x1[(((long)b * NCOL) + c * LCH + t) * DM + d];
            cnt++;
        }
    }
    // cnt is 0, 1 or 2 (overlap < stride); acc==0 when cnt==0.
    out1[idx] = acc * ((cnt == 2) ? 0.5f : 1.0f);
}

// ---------------- conv1d k=500 stride=2 pad=249 + GELU ----------------
__global__ void __launch_bounds__(896) conv_kernel(
    const float* __restrict__ xin,     // [B][SEQ1][DM]
    const float* __restrict__ cw,      // [DM][DM][KW]
    const float* __restrict__ cbias,   // [DM]
    float* __restrict__ gs)            // [B][L2LEN][DM]
{
    __shared__ float win[626];
    __shared__ float wrow[DM][KW];
    const int tl = threadIdx.x;        // 0..63  : output time slot
    const int co = threadIdx.y;        // 0..13  : output channel (wave-uniform)
    const int tid = co * 64 + tl;
    const int t0 = blockIdx.x * 64;
    const int b = blockIdx.y;
    float acc = cbias[co];
    for (int ci = 0; ci < DM; ++ci){
        for (int j = tid; j < 626; j += 896){
            int p = 2 * t0 - 249 + j;
            win[j] = (p >= 0 && p < SEQ1) ? xin[((long)b * SEQ1 + p) * DM + ci] : 0.f;
        }
        for (int j = tid; j < DM * KW; j += 896){
            int o = j / KW, k = j - o * KW;
            wrow[o][k] = cw[((long)o * DM + ci) * KW + k];
        }
        __syncthreads();
        const float* wr = wrow[co];
        const float* wi = &win[2 * tl];
        #pragma unroll 4
        for (int k = 0; k < KW; ++k) acc = fmaf(wr[k], wi[k], acc);
        __syncthreads();
    }
    int t = t0 + tl;
    if (t < L2LEN) gs[((long)b * L2LEN + t) * DM + co] = gelu_exact(acc);
}

// ---------------- stage 2: QKV projection into padded f16 layouts -------------
// qh/kh: [g=cbb*2+h][TPAD][32] f16 (head-dim padded to 32; q pre-scaled)
// vt   : [g][16][TPAD]         f16 (transposed, d-major, d padded to 16)
__global__ void qkv2_kernel(const float* __restrict__ gs, const float* __restrict__ pe,
    const float* __restrict__ win, const float* __restrict__ bin,
    _Float16* __restrict__ qh, _Float16* __restrict__ kh, _Float16* __restrict__ vt)
{
    int idx = blockIdx.x * blockDim.x + threadIdx.x;
    if (idx >= 2 * BB * LCH) return;
    int t = idx % LCH;
    int b = (idx / LCH) % BB;
    int c2 = idx / (LCH * BB);
    int sg = c2 * STRIDE1 + t;
    float xv[DM];
    #pragma unroll
    for (int d = 0; d < DM; ++d)
        xv[d] = gs[((long)b * L2LEN + sg) * DM + d] + pe[t * DM + d];
    float qv[DM], kv[DM], vv[DM];
    #pragma unroll
    for (int e = 0; e < DM; ++e){
        float aq = bin[e], ak = bin[DM + e], av = bin[2 * DM + e];
        #pragma unroll
        for (int d = 0; d < DM; ++d){
            aq = fmaf(win[e * DM + d],            xv[d], aq);
            ak = fmaf(win[(DM + e) * DM + d],     xv[d], ak);
            av = fmaf(win[(2 * DM + e) * DM + d], xv[d], av);
        }
        qv[e] = aq * INVSQRT7; kv[e] = ak; vv[e] = av;
    }
    int cbb = c2 * BB + b;
    #pragma unroll
    for (int h = 0; h < NHEAD; ++h){
        long rq = ((long)(cbb * 2 + h) * TPAD + t) * 32;
        #pragma unroll
        for (int dd = 0; dd < 32; ++dd){
            qh[rq + dd] = (dd < DH) ? (_Float16)qv[h * DH + dd] : (_Float16)0.f;
            kh[rq + dd] = (dd < DH) ? (_Float16)kv[h * DH + dd] : (_Float16)0.f;
        }
        #pragma unroll
        for (int dd = 0; dd < 16; ++dd)
            vt[((long)(cbb * 2 + h) * 16 + dd) * TPAD + t] =
                (dd < DH) ? (_Float16)vv[h * DH + dd] : (_Float16)0.f;
    }
}

// zero the padded t-tail rows (7500..7551) of qh/kh and columns of vt
__global__ void tail_zero_kernel(_Float16* __restrict__ qh, _Float16* __restrict__ kh,
                                 _Float16* __restrict__ vt){
    int idx = blockIdx.x * blockDim.x + threadIdx.x;
    if (idx >= 8 * (TPAD - LCH) * 32) return;
    int k = idx % 32;
    int r = idx / 32;
    int tt = r % (TPAD - LCH);
    int gI = r / (TPAD - LCH);
    long row = ((long)gI * TPAD + LCH + tt) * 32 + k;
    qh[row] = (_Float16)0.f;
    kh[row] = (_Float16)0.f;
    if (k < 16) vt[((long)gI * 16 + k) * TPAD + LCH + tt] = (_Float16)0.f;
}

// ---------------- stage 2: flash attention, one wave per 16-row Q tile --------
// 64 keys per iteration: 4 QK WMMAs + amortized online softmax + 2 PV WMMAs.
__global__ void __launch_bounds__(32) attn2_flash_kernel(
    const _Float16* __restrict__ qh, const _Float16* __restrict__ kh,
    const _Float16* __restrict__ vt, float* __restrict__ ob)
{
    __shared__ __align__(16) _Float16 pbuf[16][64];
    const int lane = threadIdx.x;
    const int n = lane & 15, hi = lane >> 4;
    const int s0 = blockIdx.x * 16;
    const int gI = blockIdx.y;             // cbb*2 + h
    const int h = gI & 1;
    const int cbb = gI >> 1;
    const long obase = (long)cbb * LCH * DM;
    const _Float16* khg = kh + (long)gI * TPAD * 32;
    const _Float16* vtg = vt + (long)gI * 16 * TPAD;

    const int sr = min(s0 + (lane & 15), LCH - 1);
    v16h aq = load_a_frag(qh + ((long)gI * TPAD + sr) * 32, hi);

    v8f oacc = {};
    float rmax[8], rsum[8];
    #pragma unroll
    for (int r = 0; r < 8; ++r){ rmax[r] = -1e30f; rsum[r] = 0.f; }

    for (int tb = 0; tb < LCH; tb += 64){
        const int t0g = tb + n,      t1g = tb + 16 + n;
        const int t2g = tb + 32 + n, t3g = tb + 48 + n;
        const bool ok0 = t0g < LCH, ok1 = t1g < LCH;
        const bool ok2 = t2g < LCH, ok3 = t3g < LCH;
        v16h bk0 = load_b_frag(khg + (long)t0g * 32, hi);
        v16h bk1 = load_b_frag(khg + (long)t1g * 32, hi);
        v16h bk2 = load_b_frag(khg + (long)t2g * 32, hi);
        v16h bk3 = load_b_frag(khg + (long)t3g * 32, hi);
        if (tb + 64 < LCH){
            __builtin_prefetch(khg + (long)(tb + 64 + n) * 32, 0, 0);
            __builtin_prefetch(vtg + (long)n * TPAD + tb + 64, 0, 0);
        }
        v8f z = {};
        v8f sA = __builtin_amdgcn_wmma_f32_16x16x32_f16(false, aq, false, bk0,
                                                        (short)0, z, false, false);
        v8f sB = __builtin_amdgcn_wmma_f32_16x16x32_f16(false, aq, false, bk1,
                                                        (short)0, z, false, false);
        v8f sC = __builtin_amdgcn_wmma_f32_16x16x32_f16(false, aq, false, bk2,
                                                        (short)0, z, false, false);
        v8f sD = __builtin_amdgcn_wmma_f32_16x16x32_f16(false, aq, false, bk3,
                                                        (short)0, z, false, false);
        #pragma unroll
        for (int r = 0; r < 8; ++r){
            float c0 = ok0 ? sA[r] : -1e30f;
            float c1 = ok1 ? sB[r] : -1e30f;
            float c2v = ok2 ? sC[r] : -1e30f;
            float c3 = ok3 ? sD[r] : -1e30f;
            float mrow = hred_max(fmaxf(fmaxf(c0, c1), fmaxf(c2v, c3)));
            float mnew = fmaxf(rmax[r], mrow);
            float sc = __expf(rmax[r] - mnew);
            float p0 = ok0 ? __expf(sA[r] - mnew) : 0.f;
            float p1 = ok1 ? __expf(sB[r] - mnew) : 0.f;
            float p2 = ok2 ? __expf(sC[r] - mnew) : 0.f;
            float p3 = ok3 ? __expf(sD[r] - mnew) : 0.f;
            rsum[r] = rsum[r] * sc + hred_sum((p0 + p1) + (p2 + p3));
            oacc[r] *= sc;
            rmax[r] = mnew;
            int row = r + 8 * hi;
            pbuf[row][n]      = (_Float16)p0;
            pbuf[row][16 + n] = (_Float16)p1;
            pbuf[row][32 + n] = (_Float16)p2;
            pbuf[row][48 + n] = (_Float16)p3;
        }
        asm volatile("s_wait_dscnt 0x0" ::: "memory");   // P stores -> A_p loads
        v16h ap0 = load_a_frag(&pbuf[lane & 15][0], hi);
        v16h ap1 = load_a_frag(&pbuf[lane & 15][32], hi);
        v16h bv0 = load_b_frag(vtg + (long)n * TPAD + tb, hi);
        v16h bv1 = load_b_frag(vtg + (long)n * TPAD + tb + 32, hi);
        oacc = __builtin_amdgcn_wmma_f32_16x16x32_f16(false, ap0, false, bv0,
                                                      (short)0, oacc, false, false);
        oacc = __builtin_amdgcn_wmma_f32_16x16x32_f16(false, ap1, false, bv1,
                                                      (short)0, oacc, false, false);
    }
    #pragma unroll
    for (int r = 0; r < 8; ++r){
        int s = s0 + r + 8 * hi;
        if (s < LCH && n < DH){
            float inv = (rsum[r] > 0.f) ? frcp(rsum[r]) : 0.f;
            ob[obase + (long)s * DM + h * DH + n] = oacc[r] * inv;
        }
    }
}

// ---------------- out-proj + overlap-add + residual + lin1/gelu/lin2 ----------
__global__ void final_kernel(const float* __restrict__ gs, const float* __restrict__ ob,
    const float* __restrict__ wo, const float* __restrict__ bo,
    const float* __restrict__ w1, const float* __restrict__ b1,
    const float* __restrict__ w2, const float* __restrict__ b2,
    float* __restrict__ out)
{
    int idx = blockIdx.x * blockDim.x + threadIdx.x;
    if (idx >= BB * L2LEN) return;
    int s = idx % L2LEN, b = idx / L2LEN;
    float acc[DM]; int cnt = 0;
    #pragma unroll
    for (int d = 0; d < DM; ++d) acc[d] = 0.f;
    #pragma unroll
    for (int c2 = 0; c2 < 2; ++c2){
        int t = s - c2 * STRIDE1;
        if (t >= 0 && t < LCH){
            long base = ((long)(c2 * BB + b) * LCH + t) * DM;
            float ov[DM];
            #pragma unroll
            for (int d = 0; d < DM; ++d) ov[d] = ob[base + d];
            #pragma unroll
            for (int e = 0; e < DM; ++e){
                float a = bo[e];
                #pragma unroll
                for (int d = 0; d < DM; ++d) a = fmaf(wo[e * DM + d], ov[d], a);
                acc[e] += a;
            }
            cnt++;
        }
    }
    // cnt is 0, 1 or 2; acc==0 when cnt==0.
    float inv = (cnt == 2) ? 0.5f : 1.0f;
    float gsv[DM];
    #pragma unroll
    for (int d = 0; d < DM; ++d)
        gsv[d] = gs[((long)b * L2LEN + s) * DM + d] + acc[d] * inv;
    float h1[DM];
    #pragma unroll
    for (int e = 0; e < DM; ++e){
        float a = b1[e];
        #pragma unroll
        for (int d = 0; d < DM; ++d) a = fmaf(w1[e * DM + d], gsv[d], a);
        h1[e] = gelu_exact(a);
    }
    #pragma unroll
    for (int e = 0; e < DM; ++e){
        float a = b2[e];
        #pragma unroll
        for (int d = 0; d < DM; ++d) a = fmaf(w2[e * DM + d], h1[d], a);
        out[((long)b * L2LEN + s) * DM + e] = a;
    }
}

// ---------------- launcher ----------------

extern "C" void kernel_launch(void* const* d_in, const int* in_sizes, int n_in,
                              void* d_out, int out_size, void* d_ws, size_t ws_size,
                              hipStream_t stream)
{
    const float* g          = (const float*)d_in[0];
    const float* enc_in_w   = (const float*)d_in[1];
    const float* enc_in_b   = (const float*)d_in[2];
    const float* enc_out_w  = (const float*)d_in[3];
    const float* enc_out_b  = (const float*)d_in[4];
    const float* enc_ff1_w  = (const float*)d_in[5];
    const float* enc_ff1_b  = (const float*)d_in[6];
    const float* enc_ff2_w  = (const float*)d_in[7];
    const float* enc_ff2_b  = (const float*)d_in[8];
    const float* enc_ln1_w  = (const float*)d_in[9];
    const float* enc_ln1_b  = (const float*)d_in[10];
    const float* enc_ln2_w  = (const float*)d_in[11];
    const float* enc_ln2_b  = (const float*)d_in[12];
    const float* conv_w     = (const float*)d_in[13];
    const float* conv_b     = (const float*)d_in[14];
    const float* attn_in_w  = (const float*)d_in[15];
    const float* attn_in_b  = (const float*)d_in[16];
    const float* attn_out_w = (const float*)d_in[17];
    const float* attn_out_b = (const float*)d_in[18];
    const float* lin1_w     = (const float*)d_in[19];
    const float* lin1_b     = (const float*)d_in[20];
    const float* lin2_w     = (const float*)d_in[21];
    const float* lin2_b     = (const float*)d_in[22];
    float* out = (float*)d_out;

    // workspace layout (float units; every region 16B aligned)
    float* ws   = (float*)d_ws;
    float* pe   = ws;                        // 105000
    float* x1   = pe   + 105000;             // 1050000
    float* out1 = x1   + 1050000;            // 840000
    float* gs0  = out1 + 840000;             // 420000
    float* ob   = gs0  + 420000;             // 420000
    _Float16* xh  = (_Float16*)(ob + 420000);          // MPAD*32 halves
    _Float16* qh  = xh + (long)MPAD * 32;              // 8*TPAD*32 halves
    _Float16* kh  = qh + (long)8 * TPAD * 32;
    _Float16* vt  = kh + (long)8 * TPAD * 32;          // 8*16*TPAD halves
    _Float16* w1h = vt + (long)8 * 16 * TPAD;          // 3*64*32 halves
    _Float16* w2h = w1h + NLAYER * FFD * 32;           // 3*16*64 halves

    pe_kernel<<<(LCH * DM + 255) / 256, 256, 0, stream>>>(pe);
    fold_kernel<<<(BB * NCHUNK1 * LCH * DM + 255) / 256, 256, 0, stream>>>(g, pe, x1);
    packw_kernel<<<(NLAYER * FFD * 32 + NLAYER * 16 * FFD + 255) / 256, 256, 0, stream>>>(
        enc_ff1_w, enc_ff2_w, w1h, w2h);

    for (int i = 0; i < NLAYER; ++i){
        enc_attn_kernel<<<(NCOL + 63) / 64, 64, 0, stream>>>(x1,
            enc_in_w + (long)i * 42 * DM, enc_in_b + (long)i * 42,
            enc_out_w + (long)i * DM * DM, enc_out_b + (long)i * DM,
            enc_ln1_w + (long)i * DM, enc_ln1_b + (long)i * DM);
        packx_kernel<<<(MPAD * 32 + 255) / 256, 256, 0, stream>>>(x1, xh);
        enc_ff_kernel<<<MPAD / 16, 32, 0, stream>>>(xh, x1,
            w1h + (long)i * FFD * 32, enc_ff1_b + (long)i * FFD,
            w2h + (long)i * 16 * FFD, enc_ff2_b + (long)i * DM,
            enc_ln2_w + (long)i * DM, enc_ln2_b + (long)i * DM);
    }

    unfold_kernel<<<(BB * SEQ1 * DM + 255) / 256, 256, 0, stream>>>(x1, out1);

    {
        dim3 cg((L2LEN + 63) / 64, BB);
        dim3 cb(64, DM);
        conv_kernel<<<cg, cb, 0, stream>>>(out1, conv_w, conv_b, gs0);
    }

    qkv2_kernel<<<(2 * BB * LCH + 63) / 64, 64, 0, stream>>>(gs0, pe,
        attn_in_w, attn_in_b, qh, kh, vt);
    tail_zero_kernel<<<(8 * (TPAD - LCH) * 32 + 255) / 256, 256, 0, stream>>>(qh, kh, vt);

    {
        dim3 ag((LCH + 15) / 16, 2 * BB * NHEAD);
        attn2_flash_kernel<<<ag, 32, 0, stream>>>(qh, kh, vt, ob);
    }

    final_kernel<<<(BB * L2LEN + 63) / 64, 64, 0, stream>>>(gs0, ob,
        attn_out_w, attn_out_b, lin1_w, lin1_b, lin2_w, lin2_b, out);
}
